// LlamaDecoderLayer_59889023975915
// MI455X (gfx1250) — matrix operand order repaired and checked by reading, not compiled, not run on hardware
//
#include <hip/hip_runtime.h>
#include <hip/hip_bf16.h>

typedef _Float16 hv8  __attribute__((ext_vector_type(8)));
typedef _Float16 hv16 __attribute__((ext_vector_type(16)));
typedef float    fv4  __attribute__((ext_vector_type(4)));
typedef float    fv8  __attribute__((ext_vector_type(8)));

#define WMMA_F16F32(a, b, c) \
    __builtin_amdgcn_wmma_f32_16x16x32_f16(false, (a), false, (b), (short)0, (c), false, false)

static constexpr int kHID = 2048;
static constexpr int kNH  = 16;
static constexpr int kHD  = 128;
static constexpr int kFF  = 8192;
static constexpr int kSEQ = 2048;

__device__ __forceinline__ hv16 cat8(hv8 lo, hv8 hi) {
    return __builtin_shufflevector(lo, hi, 0,1,2,3,4,5,6,7,8,9,10,11,12,13,14,15);
}

__device__ __forceinline__ hv8 cvt8(fv4 a, fv4 b) {
    hv8 r;
#pragma unroll
    for (int q = 0; q < 4; ++q) { r[q] = (_Float16)a[q]; r[4 + q] = (_Float16)b[q]; }
    return r;
}

// ---------------------------------------------------------------------------
// LayerNorm (fp32 in) -> f16 out. One block per row, 256 threads, 8 elems/thr.
// ---------------------------------------------------------------------------
__global__ __launch_bounds__(256) void ln_f16_kernel(
    const float* __restrict__ x, const float* __restrict__ w,
    const float* __restrict__ b, _Float16* __restrict__ out)
{
    __shared__ float red[16];
    const int row = blockIdx.x;
    const float* xr = x + (size_t)row * kHID;
    float vals[8];
    float s = 0.f, ss = 0.f;
#pragma unroll
    for (int i = 0; i < 8; ++i) {
        float v = xr[threadIdx.x + 256 * i];
        vals[i] = v; s += v; ss += v * v;
    }
#pragma unroll
    for (int m = 16; m >= 1; m >>= 1) {        // full wave32 reduction
        s  += __shfl_xor(s,  m, 32);
        ss += __shfl_xor(ss, m, 32);
    }
    if ((threadIdx.x & 31) == 0) {
        red[threadIdx.x >> 5]       = s;
        red[8 + (threadIdx.x >> 5)] = ss;
    }
    __syncthreads();
    float ts = 0.f, tss = 0.f;
#pragma unroll
    for (int i = 0; i < 8; ++i) { ts += red[i]; tss += red[8 + i]; }
    const float mu  = ts * (1.f / kHID);
    const float var = tss * (1.f / kHID) - mu * mu;
    const float inv = rsqrtf(var + 1e-5f);
#pragma unroll
    for (int i = 0; i < 8; ++i) {
        int c = threadIdx.x + 256 * i;
        out[(size_t)row * kHID + c] = (_Float16)((vals[i] - mu) * inv * w[c] + b[c]);
    }
}

// ---------------------------------------------------------------------------
// GEMM: out_f32[M,N] = A_f16[M,K] @ W_f32[N,K]^T (+bias) (+residual)
// Block tile 128(M) x 128(N); 8 waves as 4(M) x 2(N); wave tile 32 x 64
// (2 A-frags x 4 B-frags -> 8 WMMA / K-step). Register double-buffering of
// the global tile; all 6 fragment ds_loads issued as one burst before the
// WMMA chain; weight stream prefetched into GL2 ahead of use.
// ---------------------------------------------------------------------------
__global__ __launch_bounds__(256) void gemm_xw_kernel(
    const _Float16* __restrict__ A, const float* __restrict__ W,
    const float* __restrict__ bias, const float* __restrict__ resid,
    float* __restrict__ out, int M, int N, int K)
{
    __shared__ __align__(32) _Float16 As[128 * 32];
    __shared__ __align__(32) _Float16 Ws[128 * 32];
    const int tid  = threadIdx.x;
    const int wave = tid >> 5, lane = tid & 31;
    const int l15 = lane & 15, hiH = lane >> 4;
    const int wm = wave & 3, wn = wave >> 2;
    const int Mblk = blockIdx.x * 128, Nblk = blockIdx.y * 128;

    // staging indices
    const int ar0 = tid >> 2,           akc = (tid & 3) * 8;          // A chunk 0
    const int ar1 = (tid + 256) >> 2;                                  // A chunk 1
    const int wr_ = tid >> 1,           wkc = (tid & 1) * 16;          // W row/kc

    fv8 acc[2][4] = {};
    hv8 aReg0, aReg1;
    fv4 wReg[4];

    // prologue: prefetch tile k0 = 0
    {
        aReg0 = *(const hv8*)(&A[(size_t)(Mblk + ar0) * K + akc]);
        aReg1 = *(const hv8*)(&A[(size_t)(Mblk + ar1) * K + akc]);
        const float* src = &W[(size_t)(Nblk + wr_) * K + wkc];
#pragma unroll
        for (int q = 0; q < 4; ++q) wReg[q] = *(const fv4*)(src + 4 * q);
    }

    for (int k0 = 0; k0 < K; k0 += 32) {
        __syncthreads();
        // commit prefetched tile to LDS (f32 weights converted to f16 here)
        *(hv8*)(&As[ar0 * 32 + akc]) = aReg0;
        *(hv8*)(&As[ar1 * 32 + akc]) = aReg1;
        *(hv8*)(&Ws[wr_ * 32 + wkc])     = cvt8(wReg[0], wReg[1]);
        *(hv8*)(&Ws[wr_ * 32 + wkc + 8]) = cvt8(wReg[2], wReg[3]);
        __syncthreads();
        // prefetch next tile (loads overlap the WMMAs below)
        if (k0 + 32 < K) {
            const int kn = k0 + 32;
            aReg0 = *(const hv8*)(&A[(size_t)(Mblk + ar0) * K + kn + akc]);
            aReg1 = *(const hv8*)(&A[(size_t)(Mblk + ar1) * K + kn + akc]);
            const float* src = &W[(size_t)(Nblk + wr_) * K + kn + wkc];
#pragma unroll
            for (int q = 0; q < 4; ++q) wReg[q] = *(const fv4*)(src + 4 * q);
        }
        // deep GL2 prefetch of the streaming weight operand (global_prefetch_b8)
        if (k0 + 128 < K)
            __builtin_prefetch(&W[(size_t)(Nblk + wr_) * K + k0 + 128 + wkc], 0, 1);
        // issue ALL fragment loads as one burst, then run the 8-WMMA chain
        hv16 af[2], bf[4];
#pragma unroll
        for (int r = 0; r < 2; ++r) {
            const _Float16* ap = &As[(wm * 32 + r * 16 + l15) * 32 + hiH * 16];
            af[r] = cat8(*(const hv8*)ap, *(const hv8*)(ap + 8));
        }
#pragma unroll
        for (int j = 0; j < 4; ++j)
            bf[j] = *(const hv16*)(&Ws[(wn * 64 + j * 16 + l15) * 32 + hiH * 16]);
#pragma unroll
        for (int j = 0; j < 4; ++j) {
            acc[0][j] = WMMA_F16F32(af[0], bf[j], acc[0][j]);
            acc[1][j] = WMMA_F16F32(af[1], bf[j], acc[1][j]);
        }
    }
    const bool hasB = bias != nullptr, hasR = resid != nullptr;
#pragma unroll
    for (int r = 0; r < 2; ++r) {
        const int rowBase = Mblk + wm * 32 + r * 16 + hiH * 8;
#pragma unroll
        for (int j = 0; j < 4; ++j) {
            int col = Nblk + wn * 64 + j * 16 + l15;
            float bv = hasB ? bias[col] : 0.f;
#pragma unroll
            for (int i = 0; i < 8; ++i) {
                size_t idx = (size_t)(rowBase + i) * N + col;
                float v = acc[r][j][i] + bv;
                if (hasR) v += resid[idx];
                out[idx] = v;
            }
        }
    }
}

// ---------------------------------------------------------------------------
// Fused gate/up GEMM + SiLU(g)*u -> f16. Block tile 128(M) x 64(N);
// 8 waves as 4(M) x 2(N); wave tile 32 x 32 for each of G and U.
// ---------------------------------------------------------------------------
__global__ __launch_bounds__(256) void gemm_gateup_kernel(
    const _Float16* __restrict__ A, const float* __restrict__ Wg,
    const float* __restrict__ Wu, _Float16* __restrict__ out,
    int M, int N, int K)
{
    __shared__ __align__(32) _Float16 As[128 * 32];
    __shared__ __align__(32) _Float16 Gs[64 * 32];
    __shared__ __align__(32) _Float16 Us[64 * 32];
    const int tid  = threadIdx.x;
    const int wave = tid >> 5, lane = tid & 31;
    const int l15 = lane & 15, hiH = lane >> 4;
    const int wm = wave & 3, wn = wave >> 2;
    const int Mblk = blockIdx.x * 128, Nblk = blockIdx.y * 64;

    const int ar0 = tid >> 2, akc = (tid & 3) * 8;
    const int ar1 = (tid + 256) >> 2;
    const int gr_ = tid >> 2, gkc = (tid & 3) * 8;   // 64 rows x 32 K per matrix

    fv8 accG[2][2] = {}, accU[2][2] = {};
    hv8 aReg0, aReg1;
    fv4 gReg[2], uReg[2];

    {
        aReg0 = *(const hv8*)(&A[(size_t)(Mblk + ar0) * K + akc]);
        aReg1 = *(const hv8*)(&A[(size_t)(Mblk + ar1) * K + akc]);
        size_t off = (size_t)(Nblk + gr_) * K + gkc;
        gReg[0] = *(const fv4*)(&Wg[off]); gReg[1] = *(const fv4*)(&Wg[off] + 4);
        uReg[0] = *(const fv4*)(&Wu[off]); uReg[1] = *(const fv4*)(&Wu[off] + 4);
    }

    for (int k0 = 0; k0 < K; k0 += 32) {
        __syncthreads();
        *(hv8*)(&As[ar0 * 32 + akc]) = aReg0;
        *(hv8*)(&As[ar1 * 32 + akc]) = aReg1;
        *(hv8*)(&Gs[gr_ * 32 + gkc]) = cvt8(gReg[0], gReg[1]);
        *(hv8*)(&Us[gr_ * 32 + gkc]) = cvt8(uReg[0], uReg[1]);
        __syncthreads();
        if (k0 + 32 < K) {
            const int kn = k0 + 32;
            aReg0 = *(const hv8*)(&A[(size_t)(Mblk + ar0) * K + kn + akc]);
            aReg1 = *(const hv8*)(&A[(size_t)(Mblk + ar1) * K + kn + akc]);
            size_t off = (size_t)(Nblk + gr_) * K + kn + gkc;
            gReg[0] = *(const fv4*)(&Wg[off]); gReg[1] = *(const fv4*)(&Wg[off] + 4);
            uReg[0] = *(const fv4*)(&Wu[off]); uReg[1] = *(const fv4*)(&Wu[off] + 4);
        }
        if (k0 + 128 < K) {
            __builtin_prefetch(&Wg[(size_t)(Nblk + gr_) * K + k0 + 128 + gkc], 0, 1);
            __builtin_prefetch(&Wu[(size_t)(Nblk + gr_) * K + k0 + 128 + gkc], 0, 1);
        }
        hv16 af[2], bg[2], bu[2];
#pragma unroll
        for (int r = 0; r < 2; ++r) {
            const _Float16* ap = &As[(wm * 32 + r * 16 + l15) * 32 + hiH * 16];
            af[r] = cat8(*(const hv8*)ap, *(const hv8*)(ap + 8));
        }
#pragma unroll
        for (int j = 0; j < 2; ++j) {
            bg[j] = *(const hv16*)(&Gs[(wn * 32 + j * 16 + l15) * 32 + hiH * 16]);
            bu[j] = *(const hv16*)(&Us[(wn * 32 + j * 16 + l15) * 32 + hiH * 16]);
        }
#pragma unroll
        for (int j = 0; j < 2; ++j) {
            accG[0][j] = WMMA_F16F32(af[0], bg[j], accG[0][j]);
            accG[1][j] = WMMA_F16F32(af[1], bg[j], accG[1][j]);
            accU[0][j] = WMMA_F16F32(af[0], bu[j], accU[0][j]);
            accU[1][j] = WMMA_F16F32(af[1], bu[j], accU[1][j]);
        }
    }
#pragma unroll
    for (int r = 0; r < 2; ++r) {
        const int rowBase = Mblk + wm * 32 + r * 16 + hiH * 8;
#pragma unroll
        for (int j = 0; j < 2; ++j) {
            int col = Nblk + wn * 32 + j * 16 + l15;
#pragma unroll
            for (int i = 0; i < 8; ++i) {
                float g = accG[r][j][i], u = accU[r][j][i];
                float v = (g / (1.f + __expf(-g))) * u;   // SiLU(g) * u
                out[(size_t)(rowBase + i) * N + col] = (_Float16)v;
            }
        }
    }
}

// ---------------------------------------------------------------------------
// RoPE on q,k (+ f16 repack of q,k,v into per-head-contiguous [NH][S][HD])
// ---------------------------------------------------------------------------
__global__ __launch_bounds__(64) void rope_cvt_kernel(
    const float* __restrict__ qf, const float* __restrict__ kf,
    const float* __restrict__ vf, _Float16* __restrict__ qh,
    _Float16* __restrict__ kh, _Float16* __restrict__ vh,
    const int* __restrict__ total_seq, int S)
{
    const int s = blockIdx.x, h = blockIdx.y, i = threadIdx.x;   // i in [0,64)
    const int offset = *total_seq - S;
    const float inv_freq = __expf(-((2.0f * i) / 128.0f) * 9.210340371976184f);
    const float ang = (float)(s + offset) * inv_freq;
    float sn, cs;
    __sincosf(ang, &sn, &cs);
    const size_t src = (size_t)s * kHID + (size_t)h * kHD;
    const size_t dst = ((size_t)h * S + s) * kHD;
    float q1 = qf[src + i], q2 = qf[src + 64 + i];
    float k1 = kf[src + i], k2 = kf[src + 64 + i];
    qh[dst + i]      = (_Float16)(q1 * cs - q2 * sn);
    qh[dst + 64 + i] = (_Float16)(q2 * cs + q1 * sn);
    kh[dst + i]      = (_Float16)(k1 * cs - k2 * sn);
    kh[dst + 64 + i] = (_Float16)(k2 * cs + k1 * sn);
    vh[dst + i]      = (_Float16)vf[src + i];
    vh[dst + 64 + i] = (_Float16)vf[src + 64 + i];
}

// ---------------------------------------------------------------------------
// Causal flash attention. Grid (NH, S/128). 8 waves; each wave owns 16 query
// rows, full HD=128 output strip in registers (8 f32 fragments) with online
// softmax. 16-key tiles (register-prefetched); P re-laid out C-frag -> A-frag
// via per-wave LDS (zero-padded to K=32).
// ---------------------------------------------------------------------------
__global__ __launch_bounds__(256) void flash_attn_kernel(
    const _Float16* __restrict__ Q, const _Float16* __restrict__ Km,
    const _Float16* __restrict__ V, _Float16* __restrict__ Oh, int S)
{
    __shared__ __align__(32) _Float16 Kt[16 * 128];   // [key][d]
    __shared__ __align__(32) _Float16 Vt[128 * 32];   // [d][key], keys 16..31 = 0
    __shared__ __align__(32) _Float16 Pt[8][16 * 32]; // per-wave P, cols 16..31 = 0
    const int h = blockIdx.x, qb = blockIdx.y;
    const int tid = threadIdx.x, wave = tid >> 5, lane = tid & 31;
    const int l15 = lane & 15, hiH = lane >> 4;
    const int R = qb * 128 + wave * 16;              // wave's query-row base
    const _Float16* Qh = Q  + (size_t)h * S * kHD;
    const _Float16* Kh = Km + (size_t)h * S * kHD;
    const _Float16* Vh = V  + (size_t)h * S * kHD;

    // zero-pad the K=16..31 halves once
    for (int i = tid; i < 128 * 16; i += 256)
        Vt[(i >> 4) * 32 + 16 + (i & 15)] = (_Float16)0;
    for (int i = tid; i < 8 * 16 * 16; i += 256)
        Pt[i >> 8][((i >> 4) & 15) * 32 + 16 + (i & 15)] = (_Float16)0;

    // Q fragments for all 4 K-chunks (d = 0..127), loaded straight from HBM
    hv16 qfr[4];
    {
        const _Float16* qrow = &Qh[(size_t)(R + l15) * kHD];
#pragma unroll
        for (int c = 0; c < 4; ++c) {
            hv8 lo = *(const hv8*)(qrow + 32 * c + hiH * 16);
            hv8 hi = *(const hv8*)(qrow + 32 * c + hiH * 16 + 8);
            qfr[c] = cat8(lo, hi);
        }
    }
    float mrow[8], lrow[8];
#pragma unroll
    for (int i = 0; i < 8; ++i) { mrow[i] = -3.0e38f; lrow[i] = 0.f; }
    fv8 o[8] = {};
    const float scale = 0.08838834764831845f;        // 1/sqrt(128)

    const int stageRow = tid >> 4, stageD = (tid & 15) * 8;
    const int nTiles = (qb + 1) * 8;                 // causal: keys < (qb+1)*128
    // prefetch tile 0
    hv8 kReg = *(const hv8*)(&Kh[(size_t)stageRow * kHD + stageD]);
    hv8 vReg = *(const hv8*)(&Vh[(size_t)stageRow * kHD + stageD]);

    for (int t = 0; t < nTiles; ++t) {
        const int jj = t * 16;
        __syncthreads();
        *(hv8*)(&Kt[stageRow * 128 + stageD]) = kReg;     // [key][d] linear
#pragma unroll
        for (int q = 0; q < 8; ++q) Vt[(stageD + q) * 32 + stageRow] = vReg[q];
        __syncthreads();
        if (t + 1 < nTiles) {                             // prefetch next tile
            kReg = *(const hv8*)(&Kh[(size_t)(jj + 16 + stageRow) * kHD + stageD]);
            vReg = *(const hv8*)(&Vh[(size_t)(jj + 16 + stageRow) * kHD + stageD]);
        }
        if (jj <= R + 15) {                               // wave-uniform causal skip
            // scores: all 4 B-fragment loads issued before the WMMA chain
            hv16 kb[4];
#pragma unroll
            for (int c = 0; c < 4; ++c)
                kb[c] = *(const hv16*)(&Kt[l15 * 128 + 32 * c + hiH * 16]);
            fv8 sc = {};
#pragma unroll
            for (int c = 0; c < 4; ++c) sc = WMMA_F16F32(qfr[c], kb[c], sc);

            const int rb = R + hiH * 8;                   // lane's row group base
            const int col = jj + l15;
            float sv[8], mnew[8], ef[8];
#pragma unroll
            for (int i = 0; i < 8; ++i) {
                float v = sc[i] * scale;
                if (col > rb + i) v = -1.0e30f;           // causal mask
                sv[i] = v;
            }
#pragma unroll
            for (int i = 0; i < 8; ++i) {
                float v = sv[i];
#pragma unroll
                for (int m = 1; m <= 8; m <<= 1) v = fmaxf(v, __shfl_xor(v, m, 32));
                mnew[i] = fmaxf(mrow[i], v);
            }
#pragma unroll
            for (int i = 0; i < 8; ++i) {
                ef[i] = __expf(mrow[i] - mnew[i]);
                float p = __expf(sv[i] - mnew[i]);
                Pt[wave][(hiH * 8 + i) * 32 + l15] = (_Float16)p;
                float sum = p;
#pragma unroll
                for (int m = 1; m <= 8; m <<= 1) sum += __shfl_xor(sum, m, 32);
                lrow[i] = lrow[i] * ef[i] + sum;
                mrow[i] = mnew[i];
            }
            // rescale running output, then O += P @ V
#pragma unroll
            for (int j = 0; j < 8; ++j)
#pragma unroll
                for (int i = 0; i < 8; ++i) o[j][i] *= ef[i];
            const _Float16* pr = &Pt[wave][l15 * 32 + hiH * 16];
            hv16 pf = cat8(*(const hv8*)pr, *(const hv8*)(pr + 8));
#pragma unroll
            for (int j = 0; j < 8; ++j) {
                hv16 vfb = *(const hv16*)(&Vt[(j * 16 + l15) * 32 + hiH * 16]);
                o[j] = WMMA_F16F32(pf, vfb, o[j]);
            }
        }
    }
    // normalize + write out to [s][HID] slice of this head
#pragma unroll
    for (int j = 0; j < 8; ++j) {
#pragma unroll
        for (int i = 0; i < 8; ++i) {
            float v = o[j][i] * (1.f / lrow[i]);
            Oh[(size_t)(R + hiH * 8 + i) * kHID + h * kHD + j * 16 + l15] = (_Float16)v;
        }
    }
}

// ---------------------------------------------------------------------------
extern "C" void kernel_launch(void* const* d_in, const int* in_sizes, int n_in,
                              void* d_out, int out_size, void* d_ws, size_t ws_size,
                              hipStream_t stream)
{
    const float* hidden = (const float*)d_in[0];
    // d_in[1] = attention_mask (causal -1e9 tril; applied analytically)
    const float* ln1_w = (const float*)d_in[2];
    const float* ln1_b = (const float*)d_in[3];
    const float* ln2_w = (const float*)d_in[4];
    const float* ln2_b = (const float*)d_in[5];
    const float* wq = (const float*)d_in[6];  const float* bq = (const float*)d_in[7];
    const float* wk = (const float*)d_in[8];  const float* bk = (const float*)d_in[9];
    const float* wv = (const float*)d_in[10]; const float* bv = (const float*)d_in[11];
    const float* wo = (const float*)d_in[12]; const float* bo = (const float*)d_in[13];
    const float* w_gate = (const float*)d_in[14];
    const float* w_up   = (const float*)d_in[15];
    const float* w_down = (const float*)d_in[16];
    const int*   total_seq = (const int*)d_in[17];
    float* out = (float*)d_out;

    // workspace layout (256B aligned bumps)
    char* ws = (char*)d_ws;
    size_t off = 0;
    auto bump = [&](size_t bytes) { char* p = ws + off; off = (off + bytes + 255) & ~(size_t)255; return p; };
    _Float16* xh  = (_Float16*)bump((size_t)kSEQ * kHID * 2);  // LN1 out
    float*    qf  = (float*)   bump((size_t)kSEQ * kHID * 4);
    float*    kf  = (float*)   bump((size_t)kSEQ * kHID * 4);
    float*    vf  = (float*)   bump((size_t)kSEQ * kHID * 4);
    _Float16* qh  = (_Float16*)bump((size_t)kSEQ * kHID * 2);  // [NH][S][HD]
    _Float16* kh  = (_Float16*)bump((size_t)kSEQ * kHID * 2);
    _Float16* vh  = (_Float16*)bump((size_t)kSEQ * kHID * 2);
    _Float16* ah  = (_Float16*)bump((size_t)kSEQ * kHID * 2);  // attention out (f16)
    float*    hres= (float*)   bump((size_t)kSEQ * kHID * 4);  // residual h
    _Float16* yh  = (_Float16*)bump((size_t)kSEQ * kHID * 2);  // LN2 out
    _Float16* ffh = (_Float16*)bump((size_t)kSEQ * kFF  * 2);  // silu(g)*u
    (void)ws_size; (void)in_sizes; (void)n_in; (void)out_size;

    const dim3 blk256(256), blk64(64);
    const dim3 gProj(kSEQ / 128, kHID / 128);  // 16 x 16
    const dim3 gFF  (kSEQ / 128, kFF  / 64);   // 16 x 128

    // 1) x = LN1(hidden) -> f16
    ln_f16_kernel<<<kSEQ, blk256, 0, stream>>>(hidden, ln1_w, ln1_b, xh);
    // 2) q,k,v = x @ W^T + b
    gemm_xw_kernel<<<gProj, blk256, 0, stream>>>(xh, wq, bq, nullptr, qf, kSEQ, kHID, kHID);
    gemm_xw_kernel<<<gProj, blk256, 0, stream>>>(xh, wk, bk, nullptr, kf, kSEQ, kHID, kHID);
    gemm_xw_kernel<<<gProj, blk256, 0, stream>>>(xh, wv, bv, nullptr, vf, kSEQ, kHID, kHID);
    // 3) RoPE + per-head f16 repack
    rope_cvt_kernel<<<dim3(kSEQ, kNH), blk64, 0, stream>>>(qf, kf, vf, qh, kh, vh, total_seq, kSEQ);
    // 4) causal flash attention
    flash_attn_kernel<<<dim3(kNH, kSEQ / 128), blk256, 0, stream>>>(qh, kh, vh, ah, kSEQ);
    // 5) h = hidden + attn @ wo^T + bo
    gemm_xw_kernel<<<gProj, blk256, 0, stream>>>(ah, wo, bo, hidden, hres, kSEQ, kHID, kHID);
    // 6) y = LN2(h) -> f16
    ln_f16_kernel<<<kSEQ, blk256, 0, stream>>>(hres, ln2_w, ln2_b, yh);
    // 7) ff = silu(y@wg^T) * (y@wu^T) -> f16
    gemm_gateup_kernel<<<gFF, blk256, 0, stream>>>(yh, w_gate, w_up, ffh, kSEQ, kFF, kHID);
    // 8) out = h + ff @ w_down^T
    gemm_xw_kernel<<<gProj, blk256, 0, stream>>>(ffh, w_down, nullptr, hres, out, kSEQ, kHID, kFF);
}
